// SHattention_9663676416073
// MI455X (gfx1250) — matrix operand order repaired and checked by reading, not compiled, not run on hardware
//
#include <hip/hip_runtime.h>
#include <hip/hip_bf16.h>

typedef __attribute__((ext_vector_type(16))) _Float16 v16h;
typedef __attribute__((ext_vector_type(8)))  _Float16 v8h;
typedef __attribute__((ext_vector_type(8)))  float    v8f;

#define B_    64
#define S_    1024
#define DIN   32
#define H_    256
#define QBLK  128          // queries per 256-thread block (8 wave32 x 16)
#define KSTEP 32           // keys per flash step
#define NSTEP (S_ / KSTEP)
#define KLDS  (H_ + 8)     // padded row stride (halves) for K tile
#define VLDS  (KSTEP + 8)  // padded row stride (halves) for transposed V tile

static __device__ __forceinline__ v16h pack16(v8h lo, v8h hi8) {
    v16h r;
#pragma unroll
    for (int h = 0; h < 8; ++h) { r[h] = lo[h]; r[8 + h] = hi8[h]; }
    return r;
}

// ---------------- Kernel 1: QKV projection (f32 in -> f16 out), one WMMA per tile
__global__ __launch_bounds__(256) void qkv_proj(
    const float* __restrict__ x,
    const float* __restrict__ Wq, const float* __restrict__ bq,
    const float* __restrict__ Wk, const float* __restrict__ bk,
    const float* __restrict__ Wv, const float* __restrict__ bv,
    _Float16* __restrict__ Qh, _Float16* __restrict__ Kh, _Float16* __restrict__ Vh)
{
    const int ROWT = (B_ * S_) / 16;   // 4096
    const int COLT = H_ / 16;          // 16
    int tile = blockIdx.x * (blockDim.x >> 5) + (threadIdx.x >> 5);
    const int mat = tile / (ROWT * COLT);
    if (mat >= 3) return;
    tile %= ROWT * COLT;
    const int rowt = tile / COLT;
    const int colt = tile % COLT;

    const float* W    = (mat == 0) ? Wq : (mat == 1) ? Wk : Wv;
    const float* bias = (mat == 0) ? bq : (mat == 1) ? bk : bv;
    _Float16*    out  = (mat == 0) ? Qh : (mat == 1) ? Kh : Vh;

    const int lane = threadIdx.x & 31;
    const int l16  = lane & 15;
    const int hig  = lane >> 4;

    // A fragment: 16x32 rows of x
    const int arow  = rowt * 16 + l16;
    const int kbase = hig * 16;
    v16h a;
#pragma unroll
    for (int h = 0; h < 8; ++h) {
        a[h]     = (_Float16)x[arow * DIN + kbase + h];
        a[8 + h] = (_Float16)x[arow * DIN + kbase + 8 + h];
    }
    // B fragment: W[32][256] 32x16 column slab
    const int bcol = colt * 16 + l16;
    v16h bfr;
#pragma unroll
    for (int h = 0; h < 16; ++h)
        bfr[h] = (_Float16)W[(hig * 16 + h) * H_ + bcol];

    v8f c = {};
    c = __builtin_amdgcn_wmma_f32_16x16x32_f16(false, a, false, bfr,
                                               (short)0, c, false, false);
    const float bb = bias[bcol];
#pragma unroll
    for (int r = 0; r < 8; ++r) {
        const int row = rowt * 16 + r + hig * 8;
        out[(size_t)row * H_ + bcol] = (_Float16)(c[r] + bb);
    }
}

// ---- async global -> LDS copy of one K tile (32 keys x 256 h, padded rows)
// CDNA5 GLOBAL_LOAD_ASYNC_TO_LDS_B128: per-lane LDS byte address in VDST,
// 64-bit global address in VADDR; tracked by ASYNCcnt.
static __device__ __forceinline__ void issue_K_async(const _Float16* __restrict__ gK,
                                                     _Float16* __restrict__ ldst)
{
    const unsigned ldsBase = (unsigned)(uintptr_t)ldst;   // flat[31:0] == LDS offset
#pragma unroll
    for (int i = 0; i < 4; ++i) {
        const int idx  = threadIdx.x + i * 256;           // 1024 chunks of 16B
        const int key  = idx >> 5;                        // 32 x 16B per key row
        const int coff = (idx & 31) * 16;                 // byte offset within row
        const unsigned loff = ldsBase + (unsigned)(key * (KLDS * 2) + coff);
        const unsigned long long g =
            (unsigned long long)(uintptr_t)(gK + (size_t)key * H_) + (unsigned)coff;
        asm volatile("global_load_async_to_lds_b128 %0, %1, off"
                     :: "v"(loff), "v"(g) : "memory");
    }
}

static __device__ __forceinline__ void load_V_regs(const _Float16* __restrict__ gV,
                                                   v8h vreg[4])
{
#pragma unroll
    for (int i = 0; i < 4; ++i) {
        const int idx  = threadIdx.x + i * 256;
        const int key  = idx >> 5;
        const int hoff = (idx & 31) * 8;
        vreg[i] = *(const v8h*)(gV + (size_t)key * H_ + hoff);
    }
}

static __device__ __forceinline__ void store_V_transposed(_Float16* __restrict__ ldsVt,
                                                          const v8h vreg[4])
{
#pragma unroll
    for (int i = 0; i < 4; ++i) {
        const int idx  = threadIdx.x + i * 256;
        const int key  = idx >> 5;
        const int hoff = (idx & 31) * 8;
#pragma unroll
        for (int h = 0; h < 8; ++h)
            ldsVt[(hoff + h) * VLDS + key] = vreg[i][h];
    }
}

// ---------------- Kernel 2: flash attention, f16 WMMA, f32 accum,
// double-buffered LDS with async K copies pipelined under compute.
__global__ __launch_bounds__(256) void attn_kernel(
    const _Float16* __restrict__ Qh, const _Float16* __restrict__ Kh,
    const _Float16* __restrict__ Vh, float* __restrict__ Z)
{
    __shared__ _Float16 ldsK[2][KSTEP * KLDS];   // K tiles, row-major [key][h]
    __shared__ _Float16 ldsVt[2][H_ * VLDS];     // V tiles, transposed [h][key]
    __shared__ _Float16 ldsP[8 * 16 * KSTEP];    // per-wave P scratch

    const int b    = blockIdx.x / (S_ / QBLK);
    const int qblk = blockIdx.x % (S_ / QBLK);
    const int wid  = threadIdx.x >> 5;
    const int lane = threadIdx.x & 31;
    const int l16  = lane & 15;
    const int hig  = lane >> 4;

    const int    q0   = qblk * QBLK + wid * 16;
    const size_t base = (size_t)b * S_ * H_;

    // Q fragments: 8 chunks of 32 over H (resident in VGPRs)
    v16h qf[8];
    {
        const _Float16* qrow = Qh + base + (size_t)(q0 + l16) * H_;
#pragma unroll
        for (int c = 0; c < 8; ++c) {
            const int k0 = c * 32 + hig * 8;
            qf[c] = pack16(*(const v8h*)(qrow + k0), *(const v8h*)(qrow + k0 + 16));
        }
    }

    v8f acc[16];
    const v8f vzero = {};
#pragma unroll
    for (int t = 0; t < 16; ++t) acc[t] = vzero;
    float mrow[8], lrow[8];
#pragma unroll
    for (int j = 0; j < 8; ++j) { mrow[j] = -1e30f; lrow[j] = 0.0f; }

    // ---- pipeline prologue: tile 0
    v8h vreg[4];
    issue_K_async(Kh + base, &ldsK[0][0]);
    load_V_regs(Vh + base, vreg);
    asm volatile("s_wait_asynccnt 0" ::: "memory");
    store_V_transposed(&ldsVt[0][0], vreg);
    __syncthreads();

    for (int step = 0; step < NSTEP; ++step) {
        const int cur = step & 1;
        const int nxt = cur ^ 1;
        if (step + 1 < NSTEP) {
            const size_t nbase = base + (size_t)(step + 1) * KSTEP * H_;
            issue_K_async(Kh + nbase, &ldsK[nxt][0]);   // async, lands during compute
            load_V_regs(Vh + nbase, vreg);              // LOADcnt hidden under WMMAs
        }

        // ---- scores S = Q · K^T for 16 queries x 32 keys
        const _Float16* Kt = &ldsK[cur][0];
        v8f s0 = vzero, s1 = vzero;
#pragma unroll
        for (int c = 0; c < 8; ++c) {
            const int hb = c * 32 + hig * 8;
            v16h b0 = pack16(*(const v8h*)(&Kt[l16 * KLDS + hb]),
                             *(const v8h*)(&Kt[l16 * KLDS + hb + 16]));
            v16h b1 = pack16(*(const v8h*)(&Kt[(l16 + 16) * KLDS + hb]),
                             *(const v8h*)(&Kt[(l16 + 16) * KLDS + hb + 16]));
            s0 = __builtin_amdgcn_wmma_f32_16x16x32_f16(false, qf[c], false, b0,
                                                        (short)0, s0, false, false);
            s1 = __builtin_amdgcn_wmma_f32_16x16x32_f16(false, qf[c], false, b1,
                                                        (short)0, s1, false, false);
        }

        // ---- online softmax (C-layout: row = j + 8*hig, col = l16)
        float tmax[8], tsum[8], p0[8], p1[8];
#pragma unroll
        for (int j = 0; j < 8; ++j) tmax[j] = fmaxf(s0[j], s1[j]);
#pragma unroll
        for (int m = 1; m <= 8; m <<= 1)
#pragma unroll
            for (int j = 0; j < 8; ++j)
                tmax[j] = fmaxf(tmax[j], __shfl_xor(tmax[j], m, 32));
#pragma unroll
        for (int j = 0; j < 8; ++j) {
            const float mn = fmaxf(mrow[j], tmax[j]);
            const float sc = __expf(mrow[j] - mn);
            mrow[j] = mn;
            lrow[j] *= sc;
#pragma unroll
            for (int t = 0; t < 16; ++t) acc[t][j] *= sc;
            p0[j] = __expf(s0[j] - mn);
            p1[j] = __expf(s1[j] - mn);
            tsum[j] = p0[j] + p1[j];
        }
#pragma unroll
        for (int m = 1; m <= 8; m <<= 1)
#pragma unroll
            for (int j = 0; j < 8; ++j)
                tsum[j] += __shfl_xor(tsum[j], m, 32);
#pragma unroll
        for (int j = 0; j < 8; ++j) lrow[j] += tsum[j];

        // ---- P: C-layout -> A-layout via per-wave LDS (DS ops are wave-ordered)
        _Float16* myP = &ldsP[wid * 16 * KSTEP];
#pragma unroll
        for (int j = 0; j < 8; ++j) {
            const int row = j + hig * 8;
            myP[row * KSTEP + l16]      = (_Float16)p0[j];
            myP[row * KSTEP + l16 + 16] = (_Float16)p1[j];
        }
        __builtin_amdgcn_wave_barrier();
        const _Float16* prow = &myP[l16 * KSTEP + hig * 8];
        v16h pa = pack16(*(const v8h*)(prow), *(const v8h*)(prow + 16));

        // ---- O += P · V  (transposed V: B-fragments from two 8-half key runs per lane)
        const _Float16* Vt = &ldsVt[cur][0];
#pragma unroll
        for (int t = 0; t < 16; ++t) {
            const _Float16* vt = &Vt[(t * 16 + l16) * VLDS + hig * 8];
            v16h bf = pack16(*(const v8h*)(vt), *(const v8h*)(vt + 16));
            acc[t] = __builtin_amdgcn_wmma_f32_16x16x32_f16(false, pa, false, bf,
                                                            (short)0, acc[t], false, false);
        }

        // ---- land next tile: K via ASYNCcnt, V via transposed stores
        if (step + 1 < NSTEP) {
            asm volatile("s_wait_asynccnt 0" ::: "memory");
            store_V_transposed(&ldsVt[nxt][0], vreg);
        }
        __syncthreads();
    }

    // ---- epilogue: Z = O / (l * sqrt(H)),  sqrt(256) = 16
    float inv[8];
#pragma unroll
    for (int j = 0; j < 8; ++j) inv[j] = 1.0f / (lrow[j] * 16.0f);
#pragma unroll
    for (int t = 0; t < 16; ++t)
#pragma unroll
        for (int j = 0; j < 8; ++j) {
            const int row = q0 + j + hig * 8;
            Z[base + (size_t)row * H_ + t * 16 + l16] = acc[t][j] * inv[j];
        }
}

extern "C" void kernel_launch(void* const* d_in, const int* in_sizes, int n_in,
                              void* d_out, int out_size, void* d_ws, size_t ws_size,
                              hipStream_t stream) {
    const float* x  = (const float*)d_in[0];
    const float* Wq = (const float*)d_in[1];
    const float* bq = (const float*)d_in[2];
    const float* Wk = (const float*)d_in[3];
    const float* bk = (const float*)d_in[4];
    const float* Wv = (const float*)d_in[5];
    const float* bv = (const float*)d_in[6];
    float* Z = (float*)d_out;

    _Float16* Qh = (_Float16*)d_ws;
    _Float16* Kh = Qh + (size_t)B_ * S_ * H_;
    _Float16* Vh = Kh + (size_t)B_ * S_ * H_;

    const int tiles = 3 * ((B_ * S_) / 16) * (H_ / 16);  // 196608 wave-tiles
    qkv_proj<<<tiles / 8, 256, 0, stream>>>(x, Wq, bq, Wk, bk, Wv, bv, Qh, Kh, Vh);
    attn_kernel<<<B_ * (S_ / QBLK), 256, 0, stream>>>(Qh, Kh, Vh, Z);
}
